// SAGEMean3_19378892439727
// MI455X (gfx1250) — matrix operand order repaired and checked by reading, not compiled, not run on hardware
//
#include <hip/hip_runtime.h>
#include <hip/hip_bf16.h>
#include <stdint.h>

#define N_NODES 50000
#define N_EDGES 600000
#define IN_DIM  128
#define H_DIM   384          // 3*IN_DIM
#define OUT_DIM 1024
#define KTILES  12           // 384/32
#define NTILES  64           // 1024/16
#define MTILES  3125         // 50000/16

typedef __attribute__((ext_vector_type(16))) __bf16 v16bf;
typedef __attribute__((ext_vector_type(8)))  float  v8f;

__device__ __forceinline__ unsigned short f32_to_bf16(float f) {
  union { float f; unsigned u; } v; v.f = f;
  unsigned r = v.u + 0x7FFFu + ((v.u >> 16) & 1u);   // round-to-nearest-even
  return (unsigned short)(r >> 16);
}

// ---------------- zero scratch (sums + counts) ----------------
__global__ void zero_f4(float4* __restrict__ p, long n4) {
  long i = (long)blockIdx.x * blockDim.x + threadIdx.x;
  long stride = (long)gridDim.x * blockDim.x;
  float4 z = {0.f, 0.f, 0.f, 0.f};
  for (; i < n4; i += stride) p[i] = z;
}

// ---------------- edge scatter: one wave32 per edge ----------------
// lane handles 4 consecutive floats of the 128-dim feature.
__global__ void scatter_edges(const float* __restrict__ x,
                              const long long* __restrict__ ei,
                              float* __restrict__ sum_in,  float* __restrict__ cnt_in,
                              float* __restrict__ sum_out, float* __restrict__ cnt_out) {
  int tid  = blockIdx.x * blockDim.x + threadIdx.x;
  int e    = tid >> 5;
  int lane = tid & 31;
  if (e >= N_EDGES) return;
  int src = (int)ei[e];
  int dst = (int)ei[N_EDGES + e];

  const float4 xs = *(const float4*)(x + (long)src * IN_DIM + lane * 4);
  const float4 xd = *(const float4*)(x + (long)dst * IN_DIM + lane * 4);

  float* si = sum_in  + (long)dst * IN_DIM + lane * 4;  // messages src -> dst
  float* so = sum_out + (long)src * IN_DIM + lane * 4;  // messages dst -> src
  atomicAdd(si + 0, xs.x); atomicAdd(si + 1, xs.y);
  atomicAdd(si + 2, xs.z); atomicAdd(si + 3, xs.w);
  atomicAdd(so + 0, xd.x); atomicAdd(so + 1, xd.y);
  atomicAdd(so + 2, xd.z); atomicAdd(so + 3, xd.w);
  if (lane == 0) {
    atomicAdd(cnt_in  + dst, 1.0f);
    atomicAdd(cnt_out + src, 1.0f);
  }
}

// ---------------- build h = bf16([x | mean_in | mean_out]) ----------------
__global__ void build_h(const float* __restrict__ x,
                        const float* __restrict__ sum_in,  const float* __restrict__ cnt_in,
                        const float* __restrict__ sum_out, const float* __restrict__ cnt_out,
                        unsigned short* __restrict__ h) {
  long i = (long)blockIdx.x * blockDim.x + threadIdx.x;   // over N_NODES*IN_DIM
  if (i >= (long)N_NODES * IN_DIM) return;
  int n = (int)(i / IN_DIM);
  int f = (int)(i % IN_DIM);
  float ci = fmaxf(cnt_in[n],  1.0f);
  float co = fmaxf(cnt_out[n], 1.0f);
  unsigned short* row = h + (long)n * H_DIM;
  row[f]              = f32_to_bf16(x[i]);
  row[IN_DIM + f]     = f32_to_bf16(sum_in[i]  / ci);
  row[2 * IN_DIM + f] = f32_to_bf16(sum_out[i] / co);
}

// ---------------- pack W into per-lane WMMA B-fragment layout ----------------
// B tile (kt, nt): lane l holds column n = nt*16 + (l&15), K = kt*32 + (l<16 ? {0..7,16..23} : {8..15,24..31}),
// stored as 16 contiguous bf16 per lane -> GEMM reads it with two b128 loads.
__global__ void pack_W(const float* __restrict__ W, unsigned short* __restrict__ wp) {
  int idx = blockIdx.x * blockDim.x + threadIdx.x;        // KTILES*NTILES*32 threads
  if (idx >= KTILES * NTILES * 32) return;
  int lane = idx & 31;
  int tile = idx >> 5;                // kt*NTILES + nt
  int kt = tile / NTILES;
  int n  = (tile % NTILES) * 16 + (lane & 15);
  int kbase = kt * 32 + ((lane >> 4) << 3);
  unsigned short* dst = wp + (long)idx * 16;
#pragma unroll
  for (int j = 0; j < 16; ++j)
    dst[j] = f32_to_bf16(W[(long)(kbase + (j & 7) + ((j >> 3) << 4)) * OUT_DIM + n]);
}

// ---------------- WMMA GEMM: out = relu(h @ W + b) ----------------
// Block = 8 waves; block owns a 16-row strip; wave w owns columns [w*128, w*128+128).
// 12 k-steps x 8 n-tiles = 96 v_wmma_f32_16x16x32_bf16 per wave.
// A and B fragments are software-pipelined one tile ahead so each WMMA only
// waits on loads issued >=1 iteration earlier (partial s_wait_loadcnt).
__global__ void __launch_bounds__(256) sage_gemm(const unsigned short* __restrict__ h,
                                                 const unsigned short* __restrict__ wp,
                                                 const float* __restrict__ bias,
                                                 float* __restrict__ out) {
  const int lane    = threadIdx.x & 31;
  const int wave    = threadIdx.x >> 5;        // 0..7
  const int rowbase = blockIdx.x * 16;
  const int mrow    = rowbase + (lane & 15);
  const int rhalf   = (lane >> 4) << 3;        // 0 or 8

  // A fragment per lane: row m, K = kt*32 + (rhalf ? {8..15,24..31} : {0..7,16..23})
  const unsigned short* arow = h + (long)mrow * H_DIM + rhalf;

  // lane's 2 uint4s inside tile (kt, wave*8+nt)
  const uint4* bp = (const uint4*)wp + ((long)(wave * 8) * 32 + lane) * 2;
  const long KT_STRIDE = (long)NTILES * 64;    // uint4 per kt step (64 uint4 per nt step)

  union AB16 { uint4 u[2]; v16bf v; };

  v8f acc[8];
#pragma unroll
  for (int i = 0; i < 8; ++i) acc[i] = (v8f){0.f,0.f,0.f,0.f,0.f,0.f,0.f,0.f};

  AB16 A, Anext, Bcur, Bnext;
  A.u[0] = *(const uint4*)(arow);
  A.u[1] = *(const uint4*)(arow + 16);
  Bcur.u[0] = bp[0];
  Bcur.u[1] = bp[1];

#pragma unroll
  for (int kt = 0; kt < KTILES; ++kt) {
    if (kt + 1 < KTILES) {
      const unsigned short* an = arow + (kt + 1) * 32;
      __builtin_prefetch(an, 0, 3);                       // WGP-scope
      Anext.u[0] = *(const uint4*)(an);
      Anext.u[1] = *(const uint4*)(an + 16);
    }
    const uint4* bkt = bp + (long)kt * KT_STRIDE;
#pragma unroll
    for (int nt = 0; nt < 8; ++nt) {
      // issue next B loads before consuming current B
      if (nt + 1 < 8) {
        Bnext.u[0] = bkt[(nt + 1) * 64];
        Bnext.u[1] = bkt[(nt + 1) * 64 + 1];
      } else if (kt + 1 < KTILES) {
        Bnext.u[0] = bkt[KT_STRIDE];
        Bnext.u[1] = bkt[KT_STRIDE + 1];
      }
      acc[nt] = __builtin_amdgcn_wmma_f32_16x16x32_bf16(
          false, A.v, false, Bcur.v, (short)0, acc[nt], false, false);
      if (!(nt == 7 && kt == KTILES - 1))
        Bcur = Bnext;                                     // folds away under full unroll
    }
    if (kt + 1 < KTILES)
      A = Anext;
  }

  // Epilogue: C/D layout -> lane (l&15)=col, VGPR i = row i (+8 for lanes 16..31)
#pragma unroll
  for (int nt = 0; nt < 8; ++nt) {
    int ncol = (wave * 8 + nt) * 16 + (lane & 15);
    float bv = bias[ncol];
#pragma unroll
    for (int i = 0; i < 8; ++i) {
      int r = rowbase + rhalf + i;
      float v = acc[nt][i] + bv;
      out[(long)r * OUT_DIM + ncol] = fmaxf(v, 0.0f);
    }
  }
}

extern "C" void kernel_launch(void* const* d_in, const int* in_sizes, int n_in,
                              void* d_out, int out_size, void* d_ws, size_t ws_size,
                              hipStream_t stream) {
  const float*     x   = (const float*)d_in[0];
  const long long* ei  = (const long long*)d_in[1];  // int64 edge_index [2, E]
  const float*     W   = (const float*)d_in[2];
  const float*     b   = (const float*)d_in[3];
  float*           out = (float*)d_out;

  // Workspace layout (256B aligned slices). Sums+counts first so one zero pass covers them.
  char* ws = (char*)d_ws;
  size_t off = 0;
  auto take = [&](size_t bytes) {
    char* p = ws + off;
    off += (bytes + 255) & ~(size_t)255;
    return p;
  };
  float* sum_in  = (float*)take((size_t)N_NODES * IN_DIM * 4);   // 25.6 MB
  float* sum_out = (float*)take((size_t)N_NODES * IN_DIM * 4);   // 25.6 MB
  float* cnt_in  = (float*)take((size_t)N_NODES * 4);
  float* cnt_out = (float*)take((size_t)N_NODES * 4);
  size_t zero_bytes = off;                                        // contiguous region to zero
  unsigned short* h  = (unsigned short*)take((size_t)N_NODES * H_DIM * 2);           // 38.4 MB
  unsigned short* wp = (unsigned short*)take((size_t)KTILES * NTILES * 32 * 16 * 2); // 768 KB

  // 1) zero sums + counts
  long n4 = (long)(zero_bytes / 16);
  zero_f4<<<4096, 256, 0, stream>>>((float4*)ws, n4);

  // 2) edge scatter (wave per edge)
  {
    long threads = (long)N_EDGES * 32;
    int grid = (int)((threads + 255) / 256);
    scatter_edges<<<grid, 256, 0, stream>>>(x, ei, sum_in, cnt_in, sum_out, cnt_out);
  }

  // 3) build bf16 h = [x | mean_in | mean_out]
  {
    long threads = (long)N_NODES * IN_DIM;
    int grid = (int)((threads + 255) / 256);
    build_h<<<grid, 256, 0, stream>>>(x, sum_in, cnt_in, sum_out, cnt_out, h);
  }

  // 4) pack W into WMMA B-fragment layout (bf16)
  {
    int threads = KTILES * NTILES * 32;
    int grid = (threads + 255) / 256;
    pack_W<<<grid, 256, 0, stream>>>(W, wp);
  }

  // 5) WMMA GEMM + bias + ReLU
  sage_gemm<<<MTILES, 256, 0, stream>>>(h, wp, b, out);

  (void)in_sizes; (void)n_in; (void)out_size; (void)ws_size;
}
